// RNNRepresentationModel_74062416053002
// MI455X (gfx1250) — matrix operand order repaired and checked by reading, not compiled, not run on hardware
//
#include <hip/hip_runtime.h>

typedef __attribute__((ext_vector_type(16))) _Float16 v16h;
typedef __attribute__((ext_vector_type(8)))  _Float16 v8h;
typedef __attribute__((ext_vector_type(8)))  float    v8f;

#define HDIM 16

// Branch-free tanh on the recurrence critical path.
// gfx1250 has native V_TANH_F32 (TRANS op, co-executes with XDL WMMA).
__device__ __forceinline__ float fast_tanh(float v) {
#if __has_builtin(__builtin_amdgcn_tanhf)
    return __builtin_amdgcn_tanhf(v);
#else
    // tanh(v) = (e - 1) / (e + 1), e = exp(2v) = exp2(2*log2(e)*v).
    // Clamp: f32 tanh saturates to +-1 well before |v|=9, avoids exp overflow.
    float xc = fminf(fmaxf(v, -9.01f), 9.01f);
    float e  = __builtin_amdgcn_exp2f(2.8853900817779268f * xc);
    return (e - 1.0f) * __builtin_amdgcn_rcpf(e + 1.0f);
#endif
}

// One wave32 owns a 16-row batch tile and runs the whole T-step recurrence.
// Per step: 2 x v_wmma_f32_16x16x32_f16 (K=32 fuses input GEMM + recurrent GEMM).
__global__ __launch_bounds__(256) void rnn2_wmma_kernel(
    const float* __restrict__ x,
    const float* __restrict__ Wih0, const float* __restrict__ Whh0,
    const float* __restrict__ bih0, const float* __restrict__ bhh0,
    const float* __restrict__ Wih1, const float* __restrict__ Whh1,
    const float* __restrict__ bih1, const float* __restrict__ bhh1,
    float* __restrict__ out, int T)
{
    // Per-wave transpose staging: 8 waves x {h0,h1} x 16x16 halves = 8 KB
    __shared__ __align__(16) _Float16 sh[8][2][256];

    const int lane = threadIdx.x & 31;
    const int wave = threadIdx.x >> 5;
    const int tile = blockIdx.x * 8 + wave;   // 16-batch-row tile id
    const int b0   = tile * 16;
    const int row  = lane & 15;               // M (A) / N (B,D) index for this lane
    const int hi   = lane >> 4;               // lane group: 0 -> K/cols 0..7, 1 -> 8..15
    const int c0   = hi * 8;

    _Float16* s0 = &sh[wave][0][0];
    _Float16* s1 = &sh[wave][1][0];

    // ---- B fragments (weights), loaded once, kept in VGPRs ----
    // B = [W_ihT ; W_hhT] (32x16): elems 0..7 hold K=c0..c0+7 (W_ih[row][c0..c0+7]),
    //                              elems 8..15 hold K=16+c0..16+c0+7 (W_hh[row][c0..c0+7])
    const float* w0 = Wih0 + row * HDIM + c0;
    const float* w1 = Wih1 + row * HDIM + c0;
    const float* u0 = Whh0 + row * HDIM + c0;
    const float* u1 = Whh1 + row * HDIM + c0;
    v16h bf0, bf1;
    #pragma unroll
    for (int k = 0; k < 8; ++k) {
        bf0[k] = (_Float16)w0[k];
        bf1[k] = (_Float16)w1[k];
        bf0[8 + k] = (_Float16)u0[k];
        bf1[8 + k] = (_Float16)u1[k];
    }

    // Combined bias per output feature N=row, folded into WMMA C operand.
    const float bias0 = bih0[row] + bhh0[row];
    const float bias1 = bih1[row] + bhh1[row];
    v8f cbias0, cbias1;
    #pragma unroll
    for (int v = 0; v < 8; ++v) { cbias0[v] = bias0; cbias1[v] = bias1; }

    // Hidden-state A fragments (h[row][c0..c0+7]) as halves; init h=0.
    v8h h0f, h1f;
    #pragma unroll
    for (int k = 0; k < 8; ++k) { h0f[k] = (_Float16)0.f; h1f[k] = (_Float16)0.f; }

    // x[b, t, :]: lane loads 8 contiguous floats of row (b0+row), cols c0..c0+7.
    const float* xp = x + ((size_t)(b0 + row) * (size_t)T) * HDIM + c0;

    float4 nA = *(const float4*)(xp);
    float4 nB = *(const float4*)(xp + 4);

    float h1out[8];

    for (int t = 0; t < T; ++t) {
        float4 cA = nA, cB = nB;
        // Branchless in-bounds prefetch of next timestep's x (clamped to last t).
        {
            int tn = t + 1;
            tn = (tn < T) ? tn : (T - 1);
            const float* xn = xp + (size_t)tn * HDIM;
            nA = *(const float4*)(xn);
            nB = *(const float4*)(xn + 4);
        }

        // ---- layer 0: h0n = tanh([x_t | h0] * [W_ih0T;W_hh0T] + bias0) ----
        v16h a;
        a[0] = (_Float16)cA.x; a[1] = (_Float16)cA.y;
        a[2] = (_Float16)cA.z; a[3] = (_Float16)cA.w;
        a[4] = (_Float16)cB.x; a[5] = (_Float16)cB.y;
        a[6] = (_Float16)cB.z; a[7] = (_Float16)cB.w;
        #pragma unroll
        for (int k = 0; k < 8; ++k) a[8 + k] = h0f[k];

        v8f acc = __builtin_amdgcn_wmma_f32_16x16x32_f16(
                      false, a, false, bf0, (short)0, cbias0, false, false);

        // D element (VGPR v, this lane) = h0n[M = v + c0][N = row]; tanh in f32,
        // scatter-store f16 image h0n[m][n] at s0[m*16+n] for the transpose.
        #pragma unroll
        for (int v = 0; v < 8; ++v) {
            float hv = fast_tanh(acc[v]);
            s0[(v + c0) * HDIM + row] = (_Float16)hv;
        }
        asm volatile("s_wait_dscnt 0" ::: "memory");  // scatter-store -> gather-load (per-wave LDS in-order)
        v8h h0n = *(const v8h*)(&s0[row * HDIM + c0]);  // ds_load_b128: h0n[row][c0..c0+7]

        // ---- layer 1: h1n = tanh([h0n | h1] * [W_ih1T;W_hh1T] + bias1) ----
        v16h a1;
        #pragma unroll
        for (int k = 0; k < 8; ++k) { a1[k] = h0n[k]; a1[8 + k] = h1f[k]; }

        v8f acc1 = __builtin_amdgcn_wmma_f32_16x16x32_f16(
                       false, a1, false, bf1, (short)0, cbias1, false, false);

        #pragma unroll
        for (int v = 0; v < 8; ++v) {
            float hv = fast_tanh(acc1[v]);
            h1out[v] = hv;                               // keep f32 for final output
            s1[(v + c0) * HDIM + row] = (_Float16)hv;
        }
        asm volatile("s_wait_dscnt 0" ::: "memory");
        h1f = *(const v8h*)(&s1[row * HDIM + c0]);
        h0f = h0n;
    }

    // Final h1 (f32 tanh results of last step): D(v,lane) = h1[b0 + v + c0][row]
    float* op = out + (size_t)(b0 + c0) * HDIM + row;
    #pragma unroll
    for (int v = 0; v < 8; ++v) op[(size_t)v * HDIM] = h1out[v];
}

extern "C" void kernel_launch(void* const* d_in, const int* in_sizes, int n_in,
                              void* d_out, int out_size, void* d_ws, size_t ws_size,
                              hipStream_t stream) {
    const float* x    = (const float*)d_in[0];
    const float* Wih0 = (const float*)d_in[1];
    const float* Whh0 = (const float*)d_in[2];
    const float* bih0 = (const float*)d_in[3];
    const float* bhh0 = (const float*)d_in[4];
    const float* Wih1 = (const float*)d_in[5];
    const float* Whh1 = (const float*)d_in[6];
    const float* bih1 = (const float*)d_in[7];
    const float* bhh1 = (const float*)d_in[8];
    float* out = (float*)d_out;

    const int B = out_size / HDIM;                 // 4096
    const int T = in_sizes[0] / (B * HDIM);        // 512
    const int tiles  = B / 16;                     // 256 waves -> good latency hiding
    const int blocks = tiles / 8;                  // 8 waves (256 threads) per block

    rnn2_wmma_kernel<<<blocks, 256, 0, stream>>>(
        x, Wih0, Whh0, bih0, bhh0, Wih1, Whh1, bih1, bhh1, out, T);
}
